// RNN_65910568124594
// MI455X (gfx1250) — matrix operand order repaired and checked by reading, not compiled, not run on hardware
//
#include <hip/hip_runtime.h>
#include <hip/hip_bf16.h>

// ---------------------------------------------------------------------------
// 2-layer LSTM for MI455X (gfx1250).
//   B=32, T=512, D=512, H=1024, 4H=4096.
// Strategy:
//   * bf16 storage for activations/weights, f32 accumulate via
//     v_wmma_f32_16x16x32_bf16 (16x16 tiles, K=32 per instruction).
//   * Input projections xg = X @ Wx + b precomputed as big WMMA GEMMs.
//   * Recurrence: ONE persistent kernel per layer; 64 workgroups loop over
//     all T=512 steps with a device-scope atomic grid barrier (monotonic
//     counter, per-wave acquire spin with s_sleep backoff).
//   * Time-major layouts [T,B,*] so per-step slices are contiguous.
// ---------------------------------------------------------------------------

#define B_   32
#define T_   512
#define D_   512
#define H_   1024
#define G4H  4096
#define BT   (B_ * T_)
#define NWG  64            // workgroups in the persistent scan

typedef __attribute__((ext_vector_type(16))) __bf16 v16bf;
typedef __attribute__((ext_vector_type(8)))  float  v8f;

union BfFrag {
    uint4 u4[2];
    v16bf v;
};

__device__ __forceinline__ unsigned short f2bf(float f) {
    unsigned int u = __float_as_uint(f);
    unsigned int r = u + 0x7fffu + ((u >> 16) & 1u);   // round-to-nearest-even
    return (unsigned short)(r >> 16);
}

__device__ __forceinline__ float sigmoidf_(float x) {
    return 1.0f / (1.0f + __expf(-x));
}

// ---------------------------------------------------------------------------
// Utility: fill u32 words
// ---------------------------------------------------------------------------
__global__ void fill_u32_kernel(unsigned int* __restrict__ p, unsigned int v, int n) {
    int i = blockIdx.x * blockDim.x + threadIdx.x;
    if (i < n) p[i] = v;
}

// ---------------------------------------------------------------------------
// x [B,T,D] f32  ->  x_tb [T*B, D] bf16   (time-major)
// ---------------------------------------------------------------------------
__global__ void cast_x_kernel(const float* __restrict__ x,
                              unsigned short* __restrict__ out) {
    long idx = (long)blockIdx.x * blockDim.x + threadIdx.x;
    if (idx >= (long)B_ * T_ * D_) return;
    int d   = (int)(idx % D_);
    long r  = idx / D_;
    int t   = (int)(r % T_);
    int b   = (int)(r / T_);
    out[((long)t * B_ + b) * D_ + d] = f2bf(x[idx]);
}

// ---------------------------------------------------------------------------
// W [K, 4096] f32  ->  WT [4096, K] bf16   (LDS-tiled transpose)
// block (32,8); grid (K/32, 4096/32)
// ---------------------------------------------------------------------------
__global__ void transpose_cast_kernel(const float* __restrict__ in,
                                      unsigned short* __restrict__ out,
                                      int K) {
    __shared__ float tile[32][33];
    const int kb = blockIdx.x * 32;
    const int nb = blockIdx.y * 32;
    const int tx = threadIdx.x, ty = threadIdx.y;
#pragma unroll
    for (int i = 0; i < 4; i++)
        tile[ty + i * 8][tx] = in[(long)(kb + ty + i * 8) * G4H + nb + tx];
    __syncthreads();
#pragma unroll
    for (int i = 0; i < 4; i++)
        out[(long)(nb + ty + i * 8) * K + kb + tx] = f2bf(tile[tx][ty + i * 8]);
}

// ---------------------------------------------------------------------------
// Big GEMM: out[M,4096] = A[M,K](bf16) @ WT[4096,K](bf16)^T + bias
// grid (M/16, 8), block 256 (8 waves). Wave w: 16 rows x 64 cols.
// ---------------------------------------------------------------------------
__global__ __launch_bounds__(256) void wmma_gemm_bias_kernel(
    const unsigned short* __restrict__ A,
    const unsigned short* __restrict__ WT,
    const float* __restrict__ bias,
    float* __restrict__ out,
    int K) {
    const int lane = threadIdx.x & 31;
    const int wave = threadIdx.x >> 5;
    const int mt   = blockIdx.x;                 // M tile (16 rows)
    const int nbase = blockIdx.y * 512 + wave * 64;

    // A fragment addressing (16-bit A layout, 16x32 MxK)
    const int arow = mt * 16 + (lane & 15);
    const unsigned short* aptr = A + (long)arow * K + ((lane >> 4) * 8);

    // B fragment addressing (16-bit B layout, 32x16 KxN), WT row = out column
    const int bsel = (lane >> 4) * 8;
    const unsigned short* bp[4];
#pragma unroll
    for (int j = 0; j < 4; j++)
        bp[j] = WT + (long)(nbase + j * 16 + (lane & 15)) * K + bsel;

    v8f acc[4] = {v8f{}, v8f{}, v8f{}, v8f{}};

    for (int k0 = 0; k0 < K; k0 += 32) {
        BfFrag a;
        a.u4[0] = *reinterpret_cast<const uint4*>(aptr + k0);
        a.u4[1] = *reinterpret_cast<const uint4*>(aptr + k0 + 16);
#pragma unroll
        for (int j = 0; j < 4; j++) {
            BfFrag b;
            b.u4[0] = *reinterpret_cast<const uint4*>(bp[j] + k0);
            b.u4[1] = *reinterpret_cast<const uint4*>(bp[j] + k0 + 16);
            acc[j] = __builtin_amdgcn_wmma_f32_16x16x32_bf16(
                false, a.v, false, b.v, (short)0, acc[j], false, false);
        }
    }

    // Epilogue: C/D layout — VGPR r: (row = r + (lane>>4)*8, col = lane&15)
    const int rlo  = (lane >> 4) * 8;
    const int ccol = lane & 15;
#pragma unroll
    for (int j = 0; j < 4; j++) {
        const int nc = nbase + j * 16 + ccol;
        const float bb = bias[nc];
#pragma unroll
        for (int r = 0; r < 8; r++)
            out[(long)(mt * 16 + r + rlo) * G4H + nc] = acc[j][r] + bb;
    }
}

// ---------------------------------------------------------------------------
// Persistent LSTM scan: one launch covers all T timesteps of a layer.
//   Per step: gates = xg_t + h(t-1) @ Wh ; cell update ; grid barrier.
// grid 64 = 2 M-tiles x 32 column-groups. block 256 (8 waves).
// Wave w: gate = w&3, nsub = w>>2 -> one 16x16 tile, full K=1024 loop.
// ---------------------------------------------------------------------------
__global__ __launch_bounds__(256) void lstm_scan_kernel(
    const float* __restrict__ xg,              // [T*B, 4096] f32
    const unsigned short* __restrict__ WhT,    // [4096, 1024] bf16
    unsigned short* __restrict__ hseq,         // [T*B, 1024] bf16 (h history)
    const unsigned short* __restrict__ hzero,  // [32, 1024] bf16 zeros
    float* __restrict__ c,                     // [32, 1024] f32 (zeroed)
    float* __restrict__ out_f32,               // optional [B,T,H] f32 or null
    unsigned int* __restrict__ bar) {          // [1] u32, zeroed
    const int lane = threadIdx.x & 31;
    const int wave = threadIdx.x >> 5;
    const int mt   = blockIdx.x & 1;           // 0..1
    const int ng   = blockIdx.x >> 1;          // 0..31
    const int gate = wave & 3;
    const int nsub = wave >> 2;                // 0..1
    const int m0   = mt * 16;
    const int ncol = ng * 32 + nsub * 16;      // column in H
    const int gcol = gate * H_ + ncol;         // column in 4H

    // Per-wave constant addressing
    const long aoff = (long)(m0 + (lane & 15)) * H_ + (lane >> 4) * 8;
    const unsigned short* bptr =
        WhT + (long)(gcol + (lane & 15)) * H_ + (lane >> 4) * 8;

    __shared__ float lds[8 * 256];

    for (int t = 0; t < T_; t++) {
        const unsigned short* hp =
            (t == 0) ? hzero : (hseq + (size_t)(t - 1) * B_ * H_);
        const unsigned short* aptr = hp + aoff;

        // ---- h(t-1) @ Wh tile: 32 WMMAs over K=1024 ----
        v8f acc = v8f{};
        for (int k0 = 0; k0 < H_; k0 += 32) {
            BfFrag a, b;
            a.u4[0] = *reinterpret_cast<const uint4*>(aptr + k0);
            a.u4[1] = *reinterpret_cast<const uint4*>(aptr + k0 + 16);
            b.u4[0] = *reinterpret_cast<const uint4*>(bptr + k0);
            b.u4[1] = *reinterpret_cast<const uint4*>(bptr + k0 + 16);
            acc = __builtin_amdgcn_wmma_f32_16x16x32_bf16(
                false, a.v, false, b.v, (short)0, acc, false, false);
        }

        // ---- exchange gate tiles through LDS: [nsub*4+gate][row][col16] ----
        {
            float* myt = lds + wave * 256;
            const int rlo  = (lane >> 4) * 8;
            const int ccol = lane & 15;
#pragma unroll
            for (int r = 0; r < 8; r++) myt[(r + rlo) * 16 + ccol] = acc[r];
        }
        __syncthreads();

        // ---- cell update: 16 rows x 32 cols = 512 elems, 2 per thread ----
        const float* xg_t = xg + (size_t)t * B_ * G4H;
        unsigned short* hout = hseq + (size_t)t * B_ * H_;
        for (int e = threadIdx.x; e < 512; e += 256) {
            const int row = e >> 5;
            const int col = e & 31;
            const int ns  = col >> 4;
            const int c16 = col & 15;
            float iv = lds[(ns * 4 + 0) * 256 + row * 16 + c16];
            float fv = lds[(ns * 4 + 1) * 256 + row * 16 + c16];
            float gv = lds[(ns * 4 + 2) * 256 + row * 16 + c16];
            float ov = lds[(ns * 4 + 3) * 256 + row * 16 + c16];
            const int b    = m0 + row;
            const int hcol = ng * 32 + col;
            const float* xgp = xg_t + (long)b * G4H;
            iv += xgp[0 * H_ + hcol];
            fv += xgp[1 * H_ + hcol];
            gv += xgp[2 * H_ + hcol];
            ov += xgp[3 * H_ + hcol];
            const float cp = c[(long)b * H_ + hcol];
            const float cn = sigmoidf_(fv) * cp + sigmoidf_(iv) * tanhf(gv);
            const float hn = sigmoidf_(ov) * tanhf(cn);
            c[(long)b * H_ + hcol] = cn;
            hout[(long)b * H_ + hcol] = f2bf(hn);
            if (out_f32)
                out_f32[(long)b * (T_ * H_) + (long)t * H_ + hcol] = hn;
        }

        // ---- grid barrier (monotonic counter; target = NWG*(t+1)) ----
        __threadfence();          // all waves: flush own stores to device scope
        __syncthreads();
        if (lane == 0) {
            if (wave == 0) {
                __hip_atomic_fetch_add(bar, 1u, __ATOMIC_ACQ_REL,
                                       __HIP_MEMORY_SCOPE_AGENT);
            }
            const unsigned int target = (unsigned int)(t + 1) * NWG;
            // every wave spins with an acquire load so each SIMD/CU
            // invalidates its own L0 before re-reading peers' h(t)
            while (__hip_atomic_load(bar, __ATOMIC_ACQUIRE,
                                     __HIP_MEMORY_SCOPE_AGENT) < target) {
                __builtin_amdgcn_s_sleep(1);
            }
        }
        __syncthreads();
    }
}

// ---------------------------------------------------------------------------
// Host-side orchestration
// ---------------------------------------------------------------------------
extern "C" void kernel_launch(void* const* d_in, const int* in_sizes, int n_in,
                              void* d_out, int out_size, void* d_ws, size_t ws_size,
                              hipStream_t stream) {
    (void)in_sizes; (void)n_in; (void)out_size; (void)ws_size;
    const float* x   = (const float*)d_in[0];   // [B,T,D]
    const float* Wx0 = (const float*)d_in[1];   // [512,4096]
    const float* Wh0 = (const float*)d_in[2];   // [1024,4096]
    const float* b0  = (const float*)d_in[3];   // [4096]
    const float* Wx1 = (const float*)d_in[4];   // [1024,4096]
    const float* Wh1 = (const float*)d_in[5];   // [1024,4096]
    const float* b1  = (const float*)d_in[6];   // [4096]
    float* out = (float*)d_out;                 // [B,T,H]

    char* ws = (char*)d_ws;
    // Workspace layout (bytes):
    size_t off = 0;
    float*          xg    = (float*)(ws + off);          off += (size_t)BT * G4H * 4;  // 268.4 MB
    unsigned short* xbf   = (unsigned short*)(ws + off); off += (size_t)BT * D_ * 2;   // 16.8 MB
    unsigned short* h0seq = (unsigned short*)(ws + off); off += (size_t)BT * H_ * 2;   // 33.6 MB
    unsigned short* h1seq = (unsigned short*)(ws + off); off += (size_t)BT * H_ * 2;   // 33.6 MB
    unsigned short* Wx0T  = (unsigned short*)(ws + off); off += (size_t)G4H * D_ * 2;  // 4.2 MB
    unsigned short* Wh0T  = (unsigned short*)(ws + off); off += (size_t)G4H * H_ * 2;  // 8.4 MB
    unsigned short* Wx1T  = (unsigned short*)(ws + off); off += (size_t)G4H * H_ * 2;
    unsigned short* Wh1T  = (unsigned short*)(ws + off); off += (size_t)G4H * H_ * 2;
    unsigned short* hzero = (unsigned short*)(ws + off); off += (size_t)B_ * H_ * 2;   // 64 KB
    float*          cbuf  = (float*)(ws + off);          off += (size_t)B_ * H_ * 4;   // 128 KB
    unsigned int*   bar   = (unsigned int*)(ws + off);   off += 256;                   // barrier

    // Zero hzero + c + barrier (contiguous region)
    {
        int nwords = (int)(((size_t)B_ * H_ * 2 + (size_t)B_ * H_ * 4 + 256) / 4);
        fill_u32_kernel<<<(nwords + 255) / 256, 256, 0, stream>>>(
            (unsigned int*)hzero, 0u, nwords);
    }

    // Convert inputs
    {
        long n = (long)B_ * T_ * D_;
        cast_x_kernel<<<(unsigned)((n + 255) / 256), 256, 0, stream>>>(x, xbf);
    }
    transpose_cast_kernel<<<dim3(D_ / 32, G4H / 32), dim3(32, 8), 0, stream>>>(Wx0, Wx0T, D_);
    transpose_cast_kernel<<<dim3(H_ / 32, G4H / 32), dim3(32, 8), 0, stream>>>(Wh0, Wh0T, H_);
    transpose_cast_kernel<<<dim3(H_ / 32, G4H / 32), dim3(32, 8), 0, stream>>>(Wx1, Wx1T, H_);
    transpose_cast_kernel<<<dim3(H_ / 32, G4H / 32), dim3(32, 8), 0, stream>>>(Wh1, Wh1T, H_);

    // ---- Layer 0 ----
    wmma_gemm_bias_kernel<<<dim3(BT / 16, 8), 256, 0, stream>>>(xbf, Wx0T, b0, xg, D_);
    lstm_scan_kernel<<<NWG, 256, 0, stream>>>(xg, Wh0T, h0seq, hzero, cbuf,
                                              nullptr, bar);

    // ---- Layer 1 ----
    {
        // re-zero c + barrier (contiguous)
        int nwords = (int)(((size_t)B_ * H_ * 4 + 256) / 4);
        fill_u32_kernel<<<(nwords + 255) / 256, 256, 0, stream>>>(
            (unsigned int*)cbuf, 0u, nwords);
    }
    wmma_gemm_bias_kernel<<<dim3(BT / 16, 8), 256, 0, stream>>>(h0seq, Wx1T, b1, xg, H_);
    lstm_scan_kernel<<<NWG, 256, 0, stream>>>(xg, Wh1T, h1seq, hzero, cbuf,
                                              out, bar);
}
